// PersonalityNeighborSelector_62268435858073
// MI455X (gfx1250) — matrix operand order repaired and checked by reading, not compiled, not run on hardware
//
#include <hip/hip_runtime.h>
#include <hip/hip_bf16.h>
#include <math.h>

// Problem constants (from reference)
#define L_LAB   20000
#define U_UNL   2048
#define DIMD    256
#define NMOD    3
#define NTR     5
#define KTOP    8
#define EPSV    1e-8f

#define LTILES          (L_LAB / 16)                           // 1250
#define NSTRIPS         8
#define TILES_PER_STRIP ((LTILES + NSTRIPS - 1) / NSTRIPS)     // 157
#define UTILES          (U_UNL / 16)                           // 128
#define NPAIRS          (NTR * 16)                             // 80 (trait, u_row) pairs per u-tile
#define PAIRS_PER_LANE  3
#define NSTEPS          (NMOD * 8)                             // 24 WMMA steps per tile

typedef __bf16 bf16_t;
typedef bf16_t v16bf __attribute__((ext_vector_type(16)));
typedef bf16_t v8bf  __attribute__((ext_vector_type(8)));
typedef float  v8f   __attribute__((ext_vector_type(8)));
typedef float  v4f   __attribute__((ext_vector_type(4)));

__device__ __forceinline__ unsigned short f32_to_bf16(float f) {
  unsigned int u = __float_as_uint(f);
  u += 0x7fffu + ((u >> 16) & 1u);          // round to nearest even
  return (unsigned short)(u >> 16);
}

// ---------------------------------------------------------------------------
// Kernel 1: L2-normalize rows of [rows][256] f32 -> bf16, modality-major out:
//   in  : [l][m][256]   (row = l*NMOD + m)
//   out : [m][rows_per_mod][256]  bf16
// ---------------------------------------------------------------------------
__global__ __launch_bounds__(256) void normalize_rows_kernel(
    const float* __restrict__ in, unsigned short* __restrict__ out,
    int rows_per_mod)
{
  __shared__ float red[256];
  const int row = blockIdx.x;
  const int tid = threadIdx.x;
  const int l = row / NMOD;
  const int m = row % NMOD;

  float x = in[(size_t)row * DIMD + tid];
  red[tid] = x * x;
  __syncthreads();
  #pragma unroll
  for (int s = 128; s > 0; s >>= 1) {
    if (tid < s) red[tid] += red[tid + s];
    __syncthreads();
  }
  float scale = 1.0f / (sqrtf(red[0]) + EPSV);
  out[((size_t)m * rows_per_mod + l) * DIMD + tid] = f32_to_bf16(x * scale);
}

// ---------------------------------------------------------------------------
// Kernel 2: fused similarity GEMM (bf16 WMMA) + trait combine + streaming top-8.
// Grid: UTILES * NSTRIPS single-wave (32-thread) workgroups.
// A fragment (16x32 bf16): lane (h=lane>>4, r=lane&15) row u0+r,
//   contiguous elements [c+16h, c+16h+16)  (ISA A layout under K-permutation pi).
// B fragment (32x16 bf16): lane column l0+r, chunks [c+16h,+8) and [c+16h+8,+8)
//   (same permutation pi -> dot product exact).
// C layout: D[r + 8h, n=lane&15] in VGPR r.
// The 24 (modality, k-step) WMMAs are flattened and B is double-buffered so
// each step's loads are issued one step ahead of consumption.
// ---------------------------------------------------------------------------
__global__ __launch_bounds__(32) void topk_kernel(
    const bf16_t* __restrict__ ubf,     // [NMOD][U_UNL][DIMD]
    const bf16_t* __restrict__ lbf,     // [NMOD][L_LAB][DIMD]
    const float*  __restrict__ logits,  // [NTR][NMOD]
    float* __restrict__ cand_vals,      // [UTILES][NTR][16][NSTRIPS][KTOP]
    int*   __restrict__ cand_idx)
{
  __shared__ float sscore[NTR * 16 * 16];

  const int lane   = threadIdx.x;
  const int strip  = blockIdx.x & (NSTRIPS - 1);
  const int u_tile = blockIdx.x >> 3;
  const int h  = lane >> 4;      // lane half
  const int nn = lane & 15;      // row (A) / column (B) / N (C)

  // softmax over modalities for each trait (tiny, redundant per lane)
  float w[NTR][NMOD];
  #pragma unroll
  for (int t = 0; t < NTR; ++t) {
    float a = logits[t * NMOD + 0];
    float b = logits[t * NMOD + 1];
    float c = logits[t * NMOD + 2];
    float mx = fmaxf(a, fmaxf(b, c));
    float ea = __expf(a - mx), eb = __expf(b - mx), ec = __expf(c - mx);
    float inv = 1.0f / (ea + eb + ec);
    w[t][0] = ea * inv; w[t][1] = eb * inv; w[t][2] = ec * inv;
  }

  // A row base pointers (per modality); compiler hoists the A fragments.
  const bf16_t* arow[NMOD];
  #pragma unroll
  for (int m = 0; m < NMOD; ++m)
    arow[m] = ubf + ((size_t)m * U_UNL + u_tile * 16 + nn) * DIMD + 16 * h;

  // private sorted top-8 per owned (trait,u_row) pair
  float tv[PAIRS_PER_LANE][KTOP];
  int   ti[PAIRS_PER_LANE][KTOP];
  #pragma unroll
  for (int j = 0; j < PAIRS_PER_LANE; ++j)
    #pragma unroll
    for (int k = 0; k < KTOP; ++k) { tv[j][k] = -3.0e38f; ti[j][k] = 0; }

  const int t0 = strip * TILES_PER_STRIP;
  const int t1 = (t0 + TILES_PER_STRIP < LTILES) ? (t0 + TILES_PER_STRIP) : LTILES;

  for (int lt = t0; lt < t1; ++lt) {
    const int l0 = lt * 16;
    v8f acc[NMOD] = {};

    const bf16_t* bcol[NMOD];
    #pragma unroll
    for (int m = 0; m < NMOD; ++m)
      bcol[m] = lbf + ((size_t)m * L_LAB + l0 + nn) * DIMD + 16 * h;

    union BU { v16bf v; v8bf half[2]; };
    BU buf[2];
    // prime the pipeline: step 0 = (m=0, c=0)
    buf[0].half[0] = *(const v8bf*)(bcol[0]);
    buf[0].half[1] = *(const v8bf*)(bcol[0] + 8);

    #pragma unroll
    for (int s = 0; s < NSTEPS; ++s) {
      if (s + 1 < NSTEPS) {          // prefetch next step's B fragment
        const int mn = (s + 1) % NMOD;
        const int cn = ((s + 1) / NMOD) * 32;
        buf[(s + 1) & 1].half[0] = *(const v8bf*)(bcol[mn] + cn);
        buf[(s + 1) & 1].half[1] = *(const v8bf*)(bcol[mn] + cn + 8);
      }
      const int m = s % NMOD;
      const int c = (s / NMOD) * 32;
      v16bf a = *(const v16bf*)(arow[m] + c);
      acc[m] = __builtin_amdgcn_wmma_f32_16x16x32_bf16(
          false, a, false, buf[s & 1].v, (short)0, acc[m], false, false);
    }

    // combine modalities into per-trait scores, stage to LDS
    #pragma unroll
    for (int t = 0; t < NTR; ++t) {
      v8f st = acc[0] * w[t][0] + acc[1] * w[t][1] + acc[2] * w[t][2];
      #pragma unroll
      for (int r = 0; r < 8; ++r)
        sscore[t * 256 + (r + 8 * h) * 16 + nn] = st[r];
    }
    asm volatile("s_wait_dscnt 0" ::: "memory");

    // streaming top-8 update: 80 pairs over 32 lanes
    #pragma unroll
    for (int j = 0; j < PAIRS_PER_LANE; ++j) {
      const int p = lane + 32 * j;
      if (p < NPAIRS) {
        const int t  = p >> 4;
        const int ur = p & 15;
        const float* row = &sscore[t * 256 + ur * 16];
        for (int n = 0; n < 16; ++n) {
          float v = row[n];
          if (v > tv[j][KTOP - 1]) {
            tv[j][KTOP - 1] = v;
            ti[j][KTOP - 1] = l0 + n;
            #pragma unroll
            for (int q = KTOP - 1; q >= 1; --q) {
              if (tv[j][q] > tv[j][q - 1]) {
                float fv = tv[j][q]; tv[j][q] = tv[j][q - 1]; tv[j][q - 1] = fv;
                int   iv = ti[j][q]; ti[j][q] = ti[j][q - 1]; ti[j][q - 1] = iv;
              }
            }
          }
        }
      }
    }
    asm volatile("s_wait_dscnt 0" ::: "memory");
  }

  // write per-strip candidates
  #pragma unroll
  for (int j = 0; j < PAIRS_PER_LANE; ++j) {
    const int p = lane + 32 * j;
    if (p < NPAIRS) {
      const int t  = p >> 4;
      const int ur = p & 15;
      size_t base = ((((size_t)u_tile * NTR + t) * 16 + ur) * NSTRIPS + strip) * KTOP;
      #pragma unroll
      for (int k = 0; k < KTOP; ++k) {
        cand_vals[base + k] = tv[j][k];
        cand_idx[base + k]  = ti[j][k];
      }
    }
  }
}

// ---------------------------------------------------------------------------
// Kernel 3: merge NSTRIPS*KTOP candidates per (u,t) -> final top-8.
// Writes neighbor_traits (f32) and neighbor_indices (int64) to d_out, and an
// int32 index table for the gather kernel.
// ---------------------------------------------------------------------------
__global__ __launch_bounds__(256) void merge_topk_kernel(
    const float* __restrict__ cand_vals, const int* __restrict__ cand_idx,
    const float* __restrict__ labeled_traits,
    float* __restrict__ out_traits, long long* __restrict__ out_idx,
    int* __restrict__ final_idx)
{
  const int tid = blockIdx.x * blockDim.x + threadIdx.x;
  if (tid >= U_UNL * NTR) return;
  const int u = tid / NTR;
  const int t = tid % NTR;
  const int u_tile = u >> 4;
  const int ur     = u & 15;
  const size_t base = (((size_t)u_tile * NTR + t) * 16 + ur) * (NSTRIPS * KTOP);

  float tv[KTOP]; int ti[KTOP];
  #pragma unroll
  for (int k = 0; k < KTOP; ++k) { tv[k] = -3.0e38f; ti[k] = 0; }

  for (int i = 0; i < NSTRIPS * KTOP; ++i) {
    float v = cand_vals[base + i];
    if (v > tv[KTOP - 1]) {
      tv[KTOP - 1] = v;
      ti[KTOP - 1] = cand_idx[base + i];
      #pragma unroll
      for (int q = KTOP - 1; q >= 1; --q) {
        if (tv[q] > tv[q - 1]) {
          float fv = tv[q]; tv[q] = tv[q - 1]; tv[q - 1] = fv;
          int   iv = ti[q]; ti[q] = ti[q - 1]; ti[q - 1] = iv;
        }
      }
    }
  }

  #pragma unroll
  for (int k = 0; k < KTOP; ++k) {
    const int li = ti[k];
    out_traits[(size_t)tid * KTOP + k] = labeled_traits[(size_t)li * NTR + t];
    out_idx[(size_t)tid * KTOP + k]    = (long long)li;
    final_idx[(size_t)tid * KTOP + k]  = li;
  }
}

// ---------------------------------------------------------------------------
// Kernel 4: gather neighbor embeddings: one block per (u,t,k), 192 lanes each
// moving one 16B vector of the 3x256 f32 row. Non-temporal stores keep the
// 252 MB output stream from evicting the L2-resident labeled table.
// ---------------------------------------------------------------------------
__global__ __launch_bounds__(192) void gather_kernel(
    const float* __restrict__ labeled_embeddings,
    const int* __restrict__ final_idx,
    float* __restrict__ out_emb)
{
  const int b = blockIdx.x;                       // (u*NTR + t)*KTOP + k
  const int li = final_idx[b];
  const v4f* src = (const v4f*)(labeled_embeddings + (size_t)li * (NMOD * DIMD));
  v4f* dst = (v4f*)(out_emb + (size_t)b * (NMOD * DIMD));
  v4f v = src[threadIdx.x];
  __builtin_nontemporal_store(v, &dst[threadIdx.x]);
}

// ---------------------------------------------------------------------------
// Host launcher
// ---------------------------------------------------------------------------
static inline size_t align_up(size_t x, size_t a) { return (x + a - 1) & ~(a - 1); }

extern "C" void kernel_launch(void* const* d_in, const int* in_sizes, int n_in,
                              void* d_out, int out_size, void* d_ws, size_t ws_size,
                              hipStream_t stream) {
  (void)in_sizes; (void)n_in; (void)out_size; (void)ws_size;

  const float* labeled_embeddings   = (const float*)d_in[0]; // [20000][3][256]
  const float* labeled_traits       = (const float*)d_in[1]; // [20000][5]
  const float* unlabeled_embeddings = (const float*)d_in[2]; // [2048][3][256]
  const float* trait_logits         = (const float*)d_in[3]; // [5][3]

  // d_out partition: emb f32 | traits f32 | indices int64
  const size_t emb_elems    = (size_t)U_UNL * NTR * KTOP * NMOD * DIMD; // 62,914,560
  const size_t traits_elems = (size_t)U_UNL * NTR * KTOP;               // 81,920
  float*     out_emb    = (float*)d_out;
  float*     out_traits = out_emb + emb_elems;
  long long* out_idx    = (long long*)(out_traits + traits_elems);

  // workspace carve
  char* w = (char*)d_ws;
  size_t off = 0;
  unsigned short* lbf = (unsigned short*)(w + off);
  off = align_up(off + (size_t)NMOD * L_LAB * DIMD * sizeof(unsigned short), 256);
  unsigned short* ubf = (unsigned short*)(w + off);
  off = align_up(off + (size_t)NMOD * U_UNL * DIMD * sizeof(unsigned short), 256);
  float* cand_vals = (float*)(w + off);
  off = align_up(off + (size_t)UTILES * NTR * 16 * NSTRIPS * KTOP * sizeof(float), 256);
  int* cand_idx = (int*)(w + off);
  off = align_up(off + (size_t)UTILES * NTR * 16 * NSTRIPS * KTOP * sizeof(int), 256);
  int* final_idx = (int*)(w + off);

  // 1) normalize -> bf16 (modality-major)
  normalize_rows_kernel<<<L_LAB * NMOD, 256, 0, stream>>>(labeled_embeddings, lbf, L_LAB);
  normalize_rows_kernel<<<U_UNL * NMOD, 256, 0, stream>>>(unlabeled_embeddings, ubf, U_UNL);

  // 2) fused WMMA similarity + trait combine + streaming per-strip top-8
  topk_kernel<<<UTILES * NSTRIPS, 32, 0, stream>>>(
      (const bf16_t*)ubf, (const bf16_t*)lbf, trait_logits, cand_vals, cand_idx);

  // 3) merge strips -> final top-8; write traits + int64 indices
  merge_topk_kernel<<<(U_UNL * NTR + 255) / 256, 256, 0, stream>>>(
      cand_vals, cand_idx, labeled_traits, out_traits, out_idx, final_idx);

  // 4) gather neighbor embeddings (bandwidth-bound: ~252 MB stores)
  gather_kernel<<<U_UNL * NTR * KTOP, 192, 0, stream>>>(
      labeled_embeddings, final_idx, out_emb);
}
